// Speller_60662118089454
// MI455X (gfx1250) — matrix operand order repaired and checked by reading, not compiled, not run on hardware
//
#include <hip/hip_runtime.h>
#include <hip/hip_bf16.h>

typedef __bf16 bf16;
typedef __attribute__((ext_vector_type(16))) __bf16 v16bf;
typedef __attribute__((ext_vector_type(8)))  float  v8f;

#define B_    128
#define E_    512
#define TENC  500
#define TDEC  550
#define VOC   31
#define NBLK  64            // persistent grid size for the fused step kernel

__device__ __forceinline__ float sigmf(float x){ return 1.f/(1.f+__expf(-x)); }
__device__ __forceinline__ float bflo(unsigned u){ return __uint_as_float(u<<16); }
__device__ __forceinline__ float bfhi(unsigned u){ return __uint_as_float(u & 0xffff0000u); }

// ---------------------------------------------------------------------------
// Software grid barrier (persistent kernel): monotonic agent-scope counter.
// Counter is reset to 0 by init_state each call, so it never wraps.
// ---------------------------------------------------------------------------
__device__ __forceinline__ void grid_sync(unsigned* bar, unsigned nb)
{
    __syncthreads();
    if (threadIdx.x == 0) {
        __threadfence();                                   // release my stores
        unsigned old  = __hip_atomic_fetch_add(bar, 1u, __ATOMIC_RELAXED,
                                               __HIP_MEMORY_SCOPE_AGENT);
        unsigned need = (old / nb + 1u) * nb;
        while (__hip_atomic_load(bar, __ATOMIC_RELAXED,
                                 __HIP_MEMORY_SCOPE_AGENT) < need)
            __builtin_amdgcn_s_sleep(1);
        __threadfence();                                   // acquire others'
    }
    __syncthreads();
}

// ---------------------------------------------------------------------------
// GEMM-NT tile body: 64x64 tile of C[M,N] = A[M,K]*B[N,K]^T, K%32==0.
// 256 threads = 8 waves; each wave: 16x32 strip -> 2x v_wmma_f32_16x16x32_bf16
// per K-step. Register double-buffered: next global tile is issued while the
// WMMAs consume the LDS tile.
// ---------------------------------------------------------------------------
template<bool A_F32>
__device__ __forceinline__ void gemm_tile_body(const void* __restrict__ Ap,
        const bf16* __restrict__ Bp, int K, int m0, int n0, int tid,
        bf16 (*As)[34], bf16 (*Bs)[34], v8f& acc0, v8f& acc1)
{
    const int lane = tid & 31;
    const int w    = tid >> 5;
    const int lrow = tid >> 2;           // 0..63
    const int lcol = (tid & 3) << 3;     // 0,8,16,24
    const int wm   = (w & 3) << 4;
    const int wn   = (w >> 2) << 5;

    uint4 qa, qb; float4 fa0, fa1;
    auto loadA = [&](int k0) {
        if (A_F32) {
            const float* s = (const float*)Ap + (size_t)(m0 + lrow) * K + k0 + lcol;
            fa0 = *(const float4*)s;  fa1 = *(const float4*)(s + 4);
        } else {
            qa = *(const uint4*)((const bf16*)Ap + (size_t)(m0 + lrow) * K + k0 + lcol);
        }
    };
    auto loadB = [&](int k0) {
        qb = *(const uint4*)(Bp + (size_t)(n0 + lrow) * K + k0 + lcol);
    };

    loadA(0); loadB(0);
    for (int k0 = 0; k0 < K; k0 += 32) {
        // commit current tile to LDS
        if (A_F32) {
            As[lrow][lcol+0]=(bf16)fa0.x; As[lrow][lcol+1]=(bf16)fa0.y;
            As[lrow][lcol+2]=(bf16)fa0.z; As[lrow][lcol+3]=(bf16)fa0.w;
            As[lrow][lcol+4]=(bf16)fa1.x; As[lrow][lcol+5]=(bf16)fa1.y;
            As[lrow][lcol+6]=(bf16)fa1.z; As[lrow][lcol+7]=(bf16)fa1.w;
        } else {
            unsigned* d = (unsigned*)&As[lrow][lcol];
            d[0]=qa.x; d[1]=qa.y; d[2]=qa.z; d[3]=qa.w;
        }
        { unsigned* d = (unsigned*)&Bs[lrow][lcol];
          d[0]=qb.x; d[1]=qb.y; d[2]=qb.z; d[3]=qb.w; }
        __syncthreads();

        // kick off next tile's global loads (covered by ds+wmma latency)
        if (k0 + 32 < K) { loadA(k0 + 32); loadB(k0 + 32); }

        // A fragment (16x32): lane m = lane%16, k pattern per ISA layout
        v16bf a, b0, b1;
        const int ar = wm + (lane & 15);
        const int kb = (lane >> 4) << 3;
        #pragma unroll
        for (int i = 0; i < 8; i++) { a[i] = As[ar][kb+i]; a[8+i] = As[ar][kb+16+i]; }
        // B fragment (32x16): lane n = lane%16, k pattern per ISA layout (same as A)
        const int br = lane & 15;
        const int bk = (lane >> 4) << 3;
        #pragma unroll
        for (int i = 0; i < 8; i++) { b0[i] = Bs[wn+br][bk+i]; b0[8+i] = Bs[wn+br][bk+16+i]; b1[i] = Bs[wn+16+br][bk+i]; b1[8+i] = Bs[wn+16+br][bk+16+i]; }

        acc0 = __builtin_amdgcn_wmma_f32_16x16x32_bf16(false, a, false, b0, (short)0, acc0, false, false);
        acc1 = __builtin_amdgcn_wmma_f32_16x16x32_bf16(false, a, false, b1, (short)0, acc1, false, false);
        __syncthreads();
    }
}

template<bool OUT_BF16, bool RELU>
__device__ __forceinline__ void gemm_epilogue(const v8f& acc0, const v8f& acc1,
        void* __restrict__ Cp, const float* __restrict__ bias,
        int N, int m0, int n0, int tid)
{
    const int lane = tid & 31;
    const int w    = tid >> 5;
    const int wm   = (w & 3) << 4;
    const int wn   = (w >> 2) << 5;
    const int mrow = m0 + wm + ((lane >> 4) << 3);
    const int nc0  = n0 + wn + (lane & 15);
    #pragma unroll
    for (int r = 0; r < 8; r++) {
        float v0 = acc0[r], v1 = acc1[r];
        if (bias) { v0 += bias[nc0]; v1 += bias[nc0 + 16]; }
        if (RELU) { v0 = fmaxf(v0, 0.f); v1 = fmaxf(v1, 0.f); }
        size_t ro = (size_t)(mrow + r) * N;
        if (OUT_BF16) {
            bf16* C = (bf16*)Cp;
            C[ro + nc0] = (bf16)v0; C[ro + nc0 + 16] = (bf16)v1;
        } else {
            float* C = (float*)Cp;
            C[ro + nc0] = v0; C[ro + nc0 + 16] = v1;
        }
    }
}

// Standalone GEMM (keys/values precompute: f32 A converted in-flight, bf16 out)
template<bool A_F32, bool OUT_BF16, bool RELU>
__global__ __launch_bounds__(256) void gemm_nt(const void* __restrict__ Ap,
        const bf16* __restrict__ Bp, void* __restrict__ Cp,
        const float* __restrict__ bias, int K, int N)
{
    __shared__ bf16 As[64][34];
    __shared__ bf16 Bs[64][34];
    const int m0 = blockIdx.y << 6, n0 = blockIdx.x << 6;
    v8f a0 = {}; v8f a1 = {};
    gemm_tile_body<A_F32>(Ap, Bp, K, m0, n0, threadIdx.x, As, Bs, a0, a1);
    gemm_epilogue<OUT_BF16, RELU>(a0, a1, Cp, bias, N, m0, n0, threadIdx.x);
}

// ---------------------------------------------------------------------------
// Fused per-step kernel: 64 persistent workgroups, 7 phases, 6 grid barriers.
// ---------------------------------------------------------------------------
struct StepP {
    const bf16 *W0, *W1, *O2C, *KEYS, *VALS, *EMBB;
    const float *BIAS0, *BIAS1, *bo2c, *emb, *bcp;
    float *GATES, *HID, *C0, *C1, *H1F;
    bf16 *XH0, *XH1, *HC;
    float *raw_out, *attn_out;
    unsigned* bar;
    int t;
};

union SMem {
    struct { bf16 As[64][34]; bf16 Bs[64][34]; } g;
    struct { float hs[512]; float aw[TENC]; } a;
    struct { float hs[512]; float lg[VOC]; int bestv; } l;
};

__device__ __forceinline__ void cell_body(int i, const float* __restrict__ gates,
        float* __restrict__ c, float* __restrict__ hf,
        bf16* __restrict__ d1, int s1, int o1,
        bf16* __restrict__ d2, int s2, int o2)
{
    int b = i >> 9, j = i & 511;
    const float* g = gates + (size_t)b * 2048;
    float ig = sigmf(g[j]);
    float fg = sigmf(g[512 + j]);
    float gg = tanhf(g[1024 + j]);
    float og = sigmf(g[1536 + j]);
    float c2 = fg * c[i] + ig * gg;
    float h2 = og * tanhf(c2);
    c[i] = c2;
    if (hf) hf[i] = h2;
    bf16 hb = (bf16)h2;
    d1[(size_t)b * s1 + o1 + j] = hb;
    d2[(size_t)b * s2 + o2 + j] = hb;
}

__device__ void attn_body(int b, int tid, const StepP& P, float* hs, float* aw)
{
    hs[tid]       = P.H1F[b * 512 + tid];
    hs[256 + tid] = P.H1F[b * 512 + 256 + tid];
    __syncthreads();
    float lmax = -1e30f;
    for (int t = tid; t < TENC; t += 256) {
        const uint4* kp = (const uint4*)(P.KEYS + ((size_t)b * TENC + t) * 512);
        float s = 0.f;
        #pragma unroll 4
        for (int i = 0; i < 64; i++) {
            uint4 q = kp[i];
            int d = i * 8;
            s += bflo(q.x)*hs[d]   + bfhi(q.x)*hs[d+1]
               + bflo(q.y)*hs[d+2] + bfhi(q.y)*hs[d+3]
               + bflo(q.z)*hs[d+4] + bfhi(q.z)*hs[d+5]
               + bflo(q.w)*hs[d+6] + bfhi(q.w)*hs[d+7];
        }
        aw[t] = s;
        lmax = fmaxf(lmax, s);
    }
    __syncthreads();
    hs[tid] = lmax; __syncthreads();
    for (int s = 128; s > 0; s >>= 1) { if (tid < s) hs[tid] = fmaxf(hs[tid], hs[tid + s]); __syncthreads(); }
    float mx = hs[0];
    __syncthreads();
    float lsum = 0.f;
    for (int t = tid; t < TENC; t += 256) { float e = __expf(aw[t] - mx); aw[t] = e; lsum += e; }
    hs[tid] = lsum; __syncthreads();
    for (int s = 128; s > 0; s >>= 1) { if (tid < s) hs[tid] += hs[tid + s]; __syncthreads(); }
    float inv = 1.f / hs[0];
    for (int t = tid; t < TENC; t += 256) {
        float a2 = aw[t] * inv;
        aw[t] = a2;
        P.attn_out[((size_t)b * TDEC + P.t) * TENC + t] = a2;
    }
    __syncthreads();
    #pragma unroll
    for (int dd = 0; dd < 2; dd++) {
        int d = tid + dd * 256;
        const unsigned short* vb = (const unsigned short*)P.VALS + (size_t)b * TENC * 512 + d;
        float acc = 0.f;
        for (int t = 0; t < TENC; t++) {
            if (t + 32 < TENC) __builtin_prefetch(vb + (size_t)(t + 32) * 512, 0, 1);
            acc += aw[t] * bflo((unsigned)vb[(size_t)t * 512]);
        }
        bf16 cb = (bf16)acc;
        P.HC [(size_t)b * 1024 + 512 + d] = cb;   // o2c input  [h1 | ctx]
        P.XH0[(size_t)b * 1536 + 512 + d] = cb;   // next-step  [emb | ctx | h0]
    }
    __syncthreads();
}

__device__ void logits_body(int b, int tid, const StepP& P, float* hs, float* lg, int* bestv)
{
    hs[tid]       = P.HID[b * 512 + tid];
    hs[256 + tid] = P.HID[b * 512 + 256 + tid];
    __syncthreads();
    if (tid < VOC) {
        const float* er = P.emb + (size_t)tid * 512;
        float s = P.bcp[tid];
        for (int d = 0; d < 512; d++) s += hs[d] * er[d];
        lg[tid] = s;
        P.raw_out[((size_t)b * TDEC + P.t) * VOC + tid] = s;
    }
    __syncthreads();
    if (tid == 0) {
        int bi = 0; float bv = lg[0];
        for (int v = 1; v < VOC; v++) if (lg[v] > bv) { bv = lg[v]; bi = v; }
        *bestv = bi;
    }
    __syncthreads();
    const int sv = *bestv;
    P.XH0[(size_t)b * 1536 + tid]       = P.EMBB[(size_t)sv * 512 + tid];
    P.XH0[(size_t)b * 1536 + 256 + tid] = P.EMBB[(size_t)sv * 512 + 256 + tid];
    __syncthreads();
}

__global__ __launch_bounds__(256) void step_kernel(StepP P)
{
    __shared__ SMem sm;
    const int blk = blockIdx.x, tid = threadIdx.x;

    // A: LSTM0 gates = [emb|ctx|h0] @ [Wih0|Whh0]^T   (128x2048, K=1536)
    {
        int n0 = (blk & 31) << 6, m0 = (blk >> 5) << 6;
        v8f a0 = {}; v8f a1 = {};
        gemm_tile_body<false>(P.XH0, P.W0, 1536, m0, n0, tid, sm.g.As, sm.g.Bs, a0, a1);
        gemm_epilogue<false, false>(a0, a1, P.GATES, P.BIAS0, 2048, m0, n0, tid);
    }
    grid_sync(P.bar, NBLK);
    // B: LSTM cell 0  (h0 -> [h0|h1] lo, and -> xh0 tail for next step)
    for (int i = blk * 256 + tid; i < B_ * 512; i += NBLK * 256)
        cell_body(i, P.GATES, P.C0, nullptr, P.XH1, 1024, 0, P.XH0, 1536, 1024);
    grid_sync(P.bar, NBLK);
    // C: LSTM1 gates = [h0|h1] @ [Wih1|Whh1]^T        (128x2048, K=1024)
    {
        int n0 = (blk & 31) << 6, m0 = (blk >> 5) << 6;
        v8f a0 = {}; v8f a1 = {};
        gemm_tile_body<false>(P.XH1, P.W1, 1024, m0, n0, tid, sm.g.As, sm.g.Bs, a0, a1);
        gemm_epilogue<false, false>(a0, a1, P.GATES, P.BIAS1, 2048, m0, n0, tid);
    }
    grid_sync(P.bar, NBLK);
    // D: LSTM cell 1  (h1 -> f32 for attention, [h0|h1] hi, [h1|ctx] lo)
    for (int i = blk * 256 + tid; i < B_ * 512; i += NBLK * 256)
        cell_body(i, P.GATES, P.C1, P.H1F, P.XH1, 1024, 512, P.HC, 1024, 0);
    grid_sync(P.bar, NBLK);
    // E: attention (energy + softmax + context), 2 batch rows per WG
    for (int b = blk; b < B_; b += NBLK)
        attn_body(b, tid, P, sm.a.hs, sm.a.aw);
    grid_sync(P.bar, NBLK);
    // F: hid = relu([h1|ctx] @ w_o2c^T + b)           (128x512, K=1024)
    if (blk < 16) {
        int n0 = (blk & 7) << 6, m0 = (blk >> 3) << 6;
        v8f a0 = {}; v8f a1 = {};
        gemm_tile_body<false>(P.HC, P.O2C, 1024, m0, n0, tid, sm.g.As, sm.g.Bs, a0, a1);
        gemm_epilogue<false, true>(a0, a1, P.HID, P.bo2c, 512, m0, n0, tid);
    }
    grid_sync(P.bar, NBLK);
    // G: tied-weight logits + greedy argmax + next-step embedding
    for (int b = blk; b < B_; b += NBLK)
        logits_body(b, tid, P, sm.l.hs, sm.l.lg, &sm.l.bestv);
}

// ---------------------------------------------------------------------------
// One-time prep + state init
// ---------------------------------------------------------------------------
__global__ __launch_bounds__(256) void prep_w0(const float* __restrict__ wih0,
        const float* __restrict__ whh0, bf16* __restrict__ W0)
{
    long i = (long)blockIdx.x * 256 + threadIdx.x;          // 2048*1536
    int n = (int)(i / 1536), k = (int)(i % 1536);
    float v = (k < 1024) ? wih0[(size_t)n * 1024 + k] : whh0[(size_t)n * 512 + (k - 1024)];
    W0[i] = (bf16)v;
}
__global__ __launch_bounds__(256) void prep_w1(const float* __restrict__ wih1,
        const float* __restrict__ whh1, bf16* __restrict__ W1)
{
    long i = (long)blockIdx.x * 256 + threadIdx.x;          // 2048*1024
    int n = (int)(i / 1024), k = (int)(i % 1024);
    float v = (k < 512) ? wih1[(size_t)n * 512 + k] : whh1[(size_t)n * 512 + (k - 512)];
    W1[i] = (bf16)v;
}
__global__ __launch_bounds__(256) void prep_misc(
        const float* __restrict__ wo2c, const float* __restrict__ emb,
        const float* __restrict__ wk, const float* __restrict__ wv,
        const float* __restrict__ bih0, const float* __restrict__ bhh0,
        const float* __restrict__ bih1, const float* __restrict__ bhh1,
        bf16* __restrict__ O2C, bf16* __restrict__ EMBB,
        bf16* __restrict__ WKT, bf16* __restrict__ WVT,
        float* __restrict__ B0, float* __restrict__ B1)
{
    long i = (long)blockIdx.x * 256 + threadIdx.x;          // 512*1024
    if (i < 512L * 1024) O2C[i] = (bf16)wo2c[i];
    if (i < (long)VOC * 512) EMBB[i] = (bf16)emb[i];
    if (i < 512L * 512) {
        int d = (int)(i / 512), e = (int)(i % 512);
        WKT[i] = (bf16)wk[(size_t)e * 512 + d];
        WVT[i] = (bf16)wv[(size_t)e * 512 + d];
    }
    if (i < 2048) { B0[i] = bih0[i] + bhh0[i]; B1[i] = bih1[i] + bhh1[i]; }
}
__global__ __launch_bounds__(256) void init_state(
        bf16* __restrict__ xh0, bf16* __restrict__ xh1, bf16* __restrict__ hc,
        float* __restrict__ c0, float* __restrict__ c1, float* __restrict__ h1,
        const bf16* __restrict__ embb, unsigned* __restrict__ bar)
{
    long i = (long)blockIdx.x * 256 + threadIdx.x;          // 128*1536
    if (i == 0) *bar = 0u;                                  // reset grid barrier
    if (i < (long)B_ * 1536) {
        int j = (int)(i % 1536);
        xh0[i] = (j < 512) ? embb[j] : (bf16)0.f;           // SOS = token 0
    }
    if (i < (long)B_ * 1024) { xh1[i] = (bf16)0.f; hc[i] = (bf16)0.f; }
    if (i < (long)B_ * 512)  { c0[i] = 0.f; c1[i] = 0.f; h1[i] = 0.f; }
}

// ---------------------------------------------------------------------------
extern "C" void kernel_launch(void* const* d_in, const int* in_sizes, int n_in,
                              void* d_out, int out_size, void* d_ws, size_t ws_size,
                              hipStream_t stream)
{
    const float* enc  = (const float*)d_in[0];
    const float* emb  = (const float*)d_in[1];
    const float* wih0 = (const float*)d_in[2];
    const float* whh0 = (const float*)d_in[3];
    const float* bih0 = (const float*)d_in[4];
    const float* bhh0 = (const float*)d_in[5];
    const float* wih1 = (const float*)d_in[6];
    const float* whh1 = (const float*)d_in[7];
    const float* bih1 = (const float*)d_in[8];
    const float* bhh1 = (const float*)d_in[9];
    const float* wk   = (const float*)d_in[10];
    const float* wv   = (const float*)d_in[11];
    const float* wo2c = (const float*)d_in[12];
    const float* bo2c = (const float*)d_in[13];
    const float* bcp  = (const float*)d_in[14];

    float* out = (float*)d_out;
    const size_t RAW = (size_t)B_ * TDEC * VOC;

    char* p = (char*)d_ws;
    auto alloc = [&](size_t bytes) { void* r = (void*)p; p += (bytes + 255) & ~(size_t)255; return r; };
    bf16*  W0    = (bf16*) alloc(2048ULL * 1536 * 2);
    bf16*  W1    = (bf16*) alloc(2048ULL * 1024 * 2);
    bf16*  O2C   = (bf16*) alloc(512ULL * 1024 * 2);
    bf16*  EMBB  = (bf16*) alloc((size_t)VOC * 512 * 2);
    bf16*  WKT   = (bf16*) alloc(512ULL * 512 * 2);
    bf16*  WVT   = (bf16*) alloc(512ULL * 512 * 2);
    bf16*  KEYS  = (bf16*) alloc((size_t)B_ * TENC * 512 * 2);
    bf16*  VALS  = (bf16*) alloc((size_t)B_ * TENC * 512 * 2);
    float* BIAS0 = (float*)alloc(2048 * 4);
    float* BIAS1 = (float*)alloc(2048 * 4);
    float* GATES = (float*)alloc((size_t)B_ * 2048 * 4);
    float* HID   = (float*)alloc((size_t)B_ * 512 * 4);
    float* C0    = (float*)alloc((size_t)B_ * 512 * 4);
    float* C1    = (float*)alloc((size_t)B_ * 512 * 4);
    float* H1F   = (float*)alloc((size_t)B_ * 512 * 4);
    bf16*  XH0   = (bf16*) alloc((size_t)B_ * 1536 * 2);   // [emb | ctx | h0]
    bf16*  XH1   = (bf16*) alloc((size_t)B_ * 1024 * 2);   // [h0  | h1]
    bf16*  HC    = (bf16*) alloc((size_t)B_ * 1024 * 2);   // [h1  | ctx]
    unsigned* BAR = (unsigned*)alloc(256);

    // ---- one-time precompute ----
    prep_w0<<<(2048 * 1536) / 256, 256, 0, stream>>>(wih0, whh0, W0);
    prep_w1<<<(2048 * 1024) / 256, 256, 0, stream>>>(wih1, whh1, W1);
    prep_misc<<<(512 * 1024) / 256, 256, 0, stream>>>(wo2c, emb, wk, wv,
            bih0, bhh0, bih1, bhh1, O2C, EMBB, WKT, WVT, BIAS0, BIAS1);
    // keys/values: [64000,512] = enc(f32) @ wk/wv -> bf16 (L2-resident scan set)
    gemm_nt<true, true, false><<<dim3(8, 1000), 256, 0, stream>>>(enc, WKT, KEYS, nullptr, 512, 512);
    gemm_nt<true, true, false><<<dim3(8, 1000), 256, 0, stream>>>(enc, WVT, VALS, nullptr, 512, 512);
    init_state<<<(B_ * 1536 + 255) / 256, 256, 0, stream>>>(XH0, XH1, HC, C0, C1, H1F, EMBB, BAR);

    // ---- sequential decode: ONE fused persistent kernel per step ----
    StepP P;
    P.W0 = W0; P.W1 = W1; P.O2C = O2C; P.KEYS = KEYS; P.VALS = VALS; P.EMBB = EMBB;
    P.BIAS0 = BIAS0; P.BIAS1 = BIAS1; P.bo2c = bo2c; P.emb = emb; P.bcp = bcp;
    P.GATES = GATES; P.HID = HID; P.C0 = C0; P.C1 = C1; P.H1F = H1F;
    P.XH0 = XH0; P.XH1 = XH1; P.HC = HC;
    P.raw_out = out; P.attn_out = out + RAW; P.bar = BAR;
    for (int t = 0; t < TDEC; ++t) {
        P.t = t;
        step_kernel<<<NBLK, 256, 0, stream>>>(P);
    }
    (void)in_sizes; (void)n_in; (void)out_size; (void)ws_size;
}
